// TexRenderMulti_34789235097717
// MI455X (gfx1250) — compile-verified, matrix operand order checked by reading, not run on hardware
//
#include <hip/hip_runtime.h>
#include <stdint.h>

#define EPSF 1e-8f
#define HW_  128
#define NP_  768
#define NF_  1024
#define NB_  3
#define TW_  512

// d_out layout (flat f32): imrender[0..49151], improb[49152..65535],
// normals[65536..74751], fg[74752..91135]
#define OFF_PROB 49152
#define OFF_NRM  65536
#define OFF_FG   74752

typedef float    v2f  __attribute__((ext_vector_type(2)));
typedef float    v8f  __attribute__((ext_vector_type(8)));
typedef _Float16 v16h __attribute__((ext_vector_type(16)));

#if defined(__HIP_DEVICE_COMPILE__) && __has_builtin(__builtin_amdgcn_wmma_f32_16x16x4_f32)
#define USE_F32_WMMA 1
#else
#define USE_F32_WMMA 0
#endif

// ---- Kernel 1: transform + project points -> ws pts5 (stride 8 floats) ----
__global__ void geom_points(const float* __restrict__ pts,
                            const float* __restrict__ camrot,
                            const float* __restrict__ campos,
                            const float* __restrict__ proj,
                            float* __restrict__ pts5) {
  int tid = blockIdx.x * blockDim.x + threadIdx.x;
  if (tid >= NB_ * NP_) return;
  int b = tid / NP_;
  const float* P = pts + tid * 3;
  const float* R = camrot + b * 9;
  const float* C = campos + b * 3;
  float dx = P[0] - C[0], dy = P[1] - C[1], dz = P[2] - C[2];
  float pcx = R[0] * dx + R[1] * dy + R[2] * dz;
  float pcy = R[3] * dx + R[4] * dy + R[5] * dz;
  float pcz = R[6] * dx + R[7] * dy + R[8] * dz;
  float inv = 1.0f / (pcz + EPSF);
  float* o = pts5 + tid * 8;
  o[0] = pcx; o[1] = pcy; o[2] = pcz;
  o[3] = pcx * proj[0] * inv;
  o[4] = pcy * proj[1] * inv;
}

// ---- Kernel 2: per-face plane coefficients + normals ----
// faceRec (16 floats/face): A0,B0,C0,A1,B1,C1,z0,z1,z2,u0,v0,u1,v1,u2,v2,pad
__global__ void geom_faces(const float* __restrict__ pts5,
                           const int*   __restrict__ faces,
                           const float* __restrict__ uv,
                           float* __restrict__ faceRec,
                           float* __restrict__ outNrm) {
  int tid = blockIdx.x * blockDim.x + threadIdx.x;
  if (tid >= NB_ * NF_) return;
  int b = tid / NF_, f = tid % NF_;
  int i0 = faces[f * 3 + 0], i1 = faces[f * 3 + 1], i2 = faces[f * 3 + 2];
  const float* q0 = pts5 + (b * NP_ + i0) * 8;
  const float* q1 = pts5 + (b * NP_ + i1) * 8;
  const float* q2 = pts5 + (b * NP_ + i2) * 8;
  float e1x = q1[0] - q0[0], e1y = q1[1] - q0[1], e1z = q1[2] - q0[2];
  float e2x = q2[0] - q0[0], e2y = q2[1] - q0[1], e2z = q2[2] - q0[2];
  float nx = e1y * e2z - e1z * e2y;
  float ny = e1z * e2x - e1x * e2z;
  float nz = e1x * e2y - e1y * e2x;
  float nn = sqrtf(nx * nx + ny * ny + nz * nz) + EPSF;
  float* on = outNrm + tid * 3;
  on[0] = nx / nn; on[1] = ny / nn; on[2] = nz / nn;
  float ax = q0[3], ay = q0[4], bx = q1[3], by = q1[4], cx = q2[3], cy = q2[4];
  float d = (by - cy) * (ax - cx) + (cx - bx) * (ay - cy);
  float inv = 1.0f / (d + EPSF);
  float* r = faceRec + tid * 16;
  if (fabsf(d) > EPSF) {
    r[0] = (by - cy) * inv;
    r[1] = (cx - bx) * inv;
    r[2] = (-(by - cy) * cx - (cx - bx) * cy) * inv;
    r[3] = (cy - ay) * inv;
    r[4] = (ax - cx) * inv;
    r[5] = (-(cy - ay) * cx - (ax - cx) * cy) * inv;
  } else {  // degenerate: force l0 = l1 = -1 -> never "inside"
    r[0] = 0.0f; r[1] = 0.0f; r[2] = -1.0f;
    r[3] = 0.0f; r[4] = 0.0f; r[5] = -1.0f;
  }
  r[6] = q0[2]; r[7] = q1[2]; r[8] = q2[2];
  const float* u0 = uv + (b * NP_ + i0) * 2;
  const float* u1 = uv + (b * NP_ + i1) * 2;
  const float* u2 = uv + (b * NP_ + i2) * 2;
  r[9]  = u0[0]; r[10] = u0[1];
  r[11] = u1[0]; r[12] = u1[1];
  r[13] = u2[0]; r[14] = u2[1];
  r[15] = 0.0f;
}

// Evaluate l = A*px + B*py + C for 16 pixels x 16 faces with one WMMA.
// D layout: lane holds face n = lane%16; v8f element r holds pixel m = r + 8*(lane/16).
__device__ __forceinline__ v8f affine_eval(int h, float px, float py,
                                           float A, float B, float C) {
  v8f acc = {};
#if USE_F32_WMMA
  v2f a;  a.x  = h ? 1.0f : px;  a.y  = h ? 0.0f : py;
  v2f bb; bb.x = h ? C    : A;   bb.y = h ? 0.0f : B;
  acc = __builtin_amdgcn_wmma_f32_16x16x4_f32(false, a, false, bb,
                                              (short)0, acc, false, false);
#else
  v16h a = {}; v16h bb = {};
  if (!h) {
    a[0]  = (_Float16)px; a[1]  = (_Float16)py; a[2]  = (_Float16)1.0f;
    bb[0] = (_Float16)A;  bb[1] = (_Float16)B;  bb[2] = (_Float16)C;
  }
  acc = __builtin_amdgcn_wmma_f32_16x16x32_f16(false, a, false, bb,
                                               (short)0, acc, false, false);
#endif
  return acc;
}

// CDNA5 async global->LDS copy (ASYNCcnt-tracked), per-lane b128.
__device__ __forceinline__ void async_copy_b128(unsigned lds_addr, const void* gaddr) {
  asm volatile("global_load_async_to_lds_b128 %0, %1, off"
               :: "v"(lds_addr), "v"(gaddr) : "memory");
}
__device__ __forceinline__ void wait_async_le1() {
  asm volatile("s_wait_asynccnt 0x1" ::: "memory");
}
__device__ __forceinline__ void wait_async_le0() {
  asm volatile("s_wait_asynccnt 0x0" ::: "memory");
}

// ---- Kernel 3: rasterize + shade + composite. 1 wave = 16-pixel row strip ----
__global__ __launch_bounds__(256) void raster(const float* __restrict__ faceRec,
                                              const float* __restrict__ tex,
                                              const float* __restrict__ ts,
                                              float* __restrict__ out) {
  __shared__ float tile[2][16 * 16];                 // double-buffered face chunk
  __shared__ unsigned long long ldsKeys[8][16];
  int lane = threadIdx.x & 31;
  int wib  = threadIdx.x >> 5;
  int strip = blockIdx.x * 8 + wib;                  // 1024 strips
  int y  = strip >> 3;
  int x0 = (strip & 7) << 4;
  int h   = lane >> 4;
  int col = lane & 15;
  float px = -1.0f + (2.0f / 127.0f) * (float)(x0 + col);
  float py =  1.0f - (2.0f / 127.0f) * (float)y;

  const unsigned long long INITKEY =
      ((unsigned long long)__float_as_uint(1.0e10f)) << 32;  // z=1e10, idx=0

  float rgb[NB_][3];
  float msk[NB_];

  for (int b = 0; b < NB_; ++b) {
    const float* fr = faceRec + b * NF_ * 16;
    // prefetch chunk 0 into buffer 0 (threads 0..63 cover 64 x 16B = 1 chunk)
    if (threadIdx.x < 64)
      async_copy_b128((unsigned)(uintptr_t)&tile[0][0] + threadIdx.x * 16,
                      (const char*)fr + threadIdx.x * 16);

    unsigned long long key[8];
#pragma unroll
    for (int r = 0; r < 8; ++r) key[r] = INITKEY;

    for (int c = 0; c < NF_ / 16; ++c) {
      int cur = c & 1;
      if (c + 1 < NF_ / 16) {
        if (threadIdx.x < 64)
          async_copy_b128((unsigned)(uintptr_t)&tile[cur ^ 1][0] + threadIdx.x * 16,
                          (const char*)(fr + (c + 1) * 16 * 16) + threadIdx.x * 16);
        wait_async_le1();                  // oldest (chunk c) complete
      } else {
        wait_async_le0();
      }
      __syncthreads();                     // tile[cur] visible to all waves

      int face = c * 16 + col;             // this lane's D column
      const float* rec = &tile[cur][col * 16];
      v8f L0 = affine_eval(h, px, py, rec[0], rec[1], rec[2]);
      v8f L1 = affine_eval(h, px, py, rec[3], rec[4], rec[5]);
      float z0 = rec[6], z1 = rec[7], z2 = rec[8];
#pragma unroll
      for (int r = 0; r < 8; ++r) {
        float l0 = L0[r], l1 = L1[r];
        float l2 = 1.0f - l0 - l1;
        float z  = l0 * z0 + l1 * z1 + l2 * z2;
        bool inside = (l0 >= 0.0f) & (l1 >= 0.0f) & (l2 >= 0.0f) & (z > EPSF);
        unsigned long long cand = inside
            ? ((((unsigned long long)__float_as_uint(z)) << 32) | (unsigned)face)
            : ~0ull;
        key[r] = (cand < key[r]) ? cand : key[r];    // branchless v_min_u64
      }
      __syncthreads();                     // reads of tile[cur] done before reuse
    }
    // argmin across the 16 face-columns (16-lane halves), ties -> lowest idx
#pragma unroll
    for (int r = 0; r < 8; ++r) {
      unsigned long long k = key[r];
      for (int m = 1; m < 16; m <<= 1) {
        unsigned long long o = __shfl_xor(k, m, 32);
        if (o < k) k = o;
      }
      key[r] = k;
    }
    __syncthreads();
    if (col == 0) {
#pragma unroll
      for (int r = 0; r < 8; ++r) ldsKeys[wib][h * 8 + r] = key[r];
    }
    __syncthreads();

    float cr = 0.0f, cg = 0.0f, cb = 0.0f, mk = 0.0f;
    if (lane < 16) {
      unsigned long long k = ldsKeys[wib][lane];
      if (k != INITKEY) {
        mk = 1.0f;
        int widx = (int)(unsigned)(k & 0xFFFFFFFFull);
        const float* rec = fr + widx * 16;
        float ppx = -1.0f + (2.0f / 127.0f) * (float)(x0 + lane);
        float l0 = rec[0] * ppx + rec[1] * py + rec[2];
        float l1 = rec[3] * ppx + rec[4] * py + rec[5];
        float l2 = 1.0f - l0 - l1;
        float tu = l0 * rec[9]  + l1 * rec[11] + l2 * rec[13];
        float tv = l0 * rec[10] + l1 * rec[12] + l2 * rec[14];
        tu = fminf(fmaxf(tu, 0.0f), 1.0f);
        tv = fminf(fmaxf(tv, 0.0f), 1.0f);
        int txi = (int)rintf(tu * (float)(TW_ - 1));            // round half-even
        int tyi = (int)rintf((1.0f - tv) * (float)(TW_ - 1));
        const float* T = tex + (size_t)b * 3 * TW_ * TW_;
        cr = T[                tyi * TW_ + txi];
        cg = T[TW_ * TW_     + tyi * TW_ + txi];
        cb = T[2 * TW_ * TW_ + tyi * TW_ + txi];
      }
    }
    rgb[b][0] = cr * mk; rgb[b][1] = cg * mk; rgb[b][2] = cb * mk;
    msk[b] = mk;
  }

  if (lane < 16) {
    // stable argsort of -ts[:,2] (descending, ties -> lower index first)
    float tz[3] = { ts[2], ts[5], ts[8] };
    int order[3];
#pragma unroll
    for (int i = 0; i < 3; ++i) {
      int pos = 0;
      for (int j = 0; j < 3; ++j)
        if ((tz[j] > tz[i]) || (tz[j] == tz[i] && j < i)) pos++;
      order[pos] = i;
    }
    int s = order[0];
    float ir = rgb[s][0], ig = rgb[s][1], ib = rgb[s][2];
    float pr = msk[s], fg = msk[s];
    for (int k = 1; k < 3; ++k) {
      s = order[k];
      if (msk[s] > 0.5f) {
        ir = rgb[s][0]; ig = rgb[s][1]; ib = rgb[s][2];
        pr = msk[s]; fg = msk[s];
      }
    }
    int pix = y * HW_ + (x0 + lane);
    out[pix * 3 + 0] = ir;
    out[pix * 3 + 1] = ig;
    out[pix * 3 + 2] = ib;
    out[OFF_PROB + pix] = pr;
    out[OFF_FG + pix]   = fg;
  }
}

extern "C" void kernel_launch(void* const* d_in, const int* in_sizes, int n_in,
                              void* d_out, int out_size, void* d_ws, size_t ws_size,
                              hipStream_t stream) {
  const float* points = (const float*)d_in[0];
  const float* camrot = (const float*)d_in[1];
  const float* campos = (const float*)d_in[2];
  const float* proj   = (const float*)d_in[3];
  const float* uv     = (const float*)d_in[4];
  const float* tex    = (const float*)d_in[5];
  const float* ts     = (const float*)d_in[6];
  const int*   faces  = (const int*)d_in[7];
  float* out = (float*)d_out;

  float* pts5    = (float*)d_ws;           // 3*768*8  floats =  72 KB
  float* faceRec = pts5 + NB_ * NP_ * 8;   // 3*1024*16 floats = 192 KB

  geom_points<<<(NB_ * NP_ + 255) / 256, 256, 0, stream>>>(points, camrot, campos,
                                                           proj, pts5);
  geom_faces<<<(NB_ * NF_ + 255) / 256, 256, 0, stream>>>(pts5, faces, uv, faceRec,
                                                          out + OFF_NRM);
  raster<<<128, 256, 0, stream>>>(faceRec, tex, ts, out);
}
